// HeteroGNN_66374424592772
// MI455X (gfx1250) — matrix operand order repaired and checked by reading, not compiled, not run on hardware
//
#include <hip/hip_runtime.h>
#include <cmath>
#include <cstdint>
#include <cstddef>

// ======================== types / constants ========================
typedef _Float16 half_t;
typedef __attribute__((ext_vector_type(16))) _Float16 v16h;
typedef __attribute__((ext_vector_type(8)))  float    v8f;

#define HID    32
#define NREL   9
#define NLAYER 3

struct WPtrs { const float* p[NLAYER * NREL * 4]; }; // 108 pointers (864 B kernarg)

// up to 4 fused outputs sharing one A operand
struct LinArgs {
  const half_t* Wt[4];
  const float*  bias[4];
  float*        Y[4];
};

// ======================== device helpers ========================
__device__ __forceinline__ float atomicMaxF(float* addr, float val) {
  // monotonic int-punning trick: works for mixed-sign floats, init with -inf
  if (val >= 0.f)
    return __int_as_float(atomicMax((int*)addr, __float_as_int(val)));
  return __uint_as_float(atomicMin((unsigned int*)addr, __float_as_uint(val)));
}
__device__ __forceinline__ float sigm(float x) { return 1.f / (1.f + __expf(-x)); }

// ======================== kernels ========================

// Transpose + convert one 32x32 fp32 weight matrix to f16: Wt[c][k] = W[k][c]
__global__ __launch_bounds__(1024) void convert_weights_kernel(WPtrs wp, half_t* __restrict__ out) {
  int mat = blockIdx.x;
  int i   = threadIdx.x;            // 0..1023
  int c   = i >> 5, kk = i & 31;
  out[(size_t)mat * HID * HID + c * HID + kk] = (half_t)wp.p[mat][kk * HID + c];
}

__global__ void fill_kernel(float* __restrict__ p, float v, int n) {
  int i = blockIdx.x * blockDim.x + threadIdx.x;
  if (i < n) p[i] = v;
}

// Fused multi-output linear: Y_o[N x 32] = X[N x 32] @ W_o[32x32] + bias_o, o < NOUT.
// A operand (converted to f16) is loaded ONCE and reused for all outputs (2 WMMAs each).
// ACCMASK bit o selects Y_o += result. Wt_o is f16 transposed: Wt[c*32+k] = W[k][c].
// Full tiles: b128 A loads, LDS-staged contiguous b128 D stores. Tail tile: guarded scalars.
template <int NOUT, int ACCMASK>
__global__ __launch_bounds__(256) void linear32_wmma_kernel(
    const float* __restrict__ X, LinArgs args, int N)
{
  __shared__ float stage[8][16 * HID];          // 2KB per wave, reused across outputs
  int wid  = threadIdx.x >> 5;
  int lane = threadIdx.x & 31;
  int wave = (int)(blockIdx.x * 8u + wid);
  int r0   = wave * 16;
  if (r0 >= N) return;                           // wave-uniform -> EXEC all-1 for WMMA

  int half_id = lane >> 4;                       // K-half select
  int ln      = lane & 15;                       // M (for A) / N (for B,C,D)
  int row     = r0 + ln;
  bool full   = (r0 + 16) <= N;                  // wave-uniform

  // A fragment: element j <-> K=16h+j (contiguous, matches B)
  v16h a;
  if (full) {
    const float4* x4 = (const float4*)(X + (size_t)row * HID);
    float4 p0 = x4[half_id * 4];
    float4 p1 = x4[half_id * 4 + 1];
    float4 p2 = x4[half_id * 4 + 2];
    float4 p3 = x4[half_id * 4 + 3];
    a[0]  = (half_t)p0.x; a[1]  = (half_t)p0.y; a[2]  = (half_t)p0.z; a[3]  = (half_t)p0.w;
    a[4]  = (half_t)p1.x; a[5]  = (half_t)p1.y; a[6]  = (half_t)p1.z; a[7]  = (half_t)p1.w;
    a[8]  = (half_t)p2.x; a[9]  = (half_t)p2.y; a[10] = (half_t)p2.z; a[11] = (half_t)p2.w;
    a[12] = (half_t)p3.x; a[13] = (half_t)p3.y; a[14] = (half_t)p3.z; a[15] = (half_t)p3.w;
  } else {
    bool in = row < N;
    const float* xr = X + (size_t)row * HID;
    #pragma unroll
    for (int j = 0; j < 8; ++j) {
      a[j]     = in ? (half_t)xr[16 * half_id + j]     : (half_t)0.f;
      a[8 + j] = in ? (half_t)xr[16 * half_id + 8 + j] : (half_t)0.f;
    }
  }

  float* tl = stage[wid];
  #pragma unroll
  for (int o = 0; o < NOUT; ++o) {
    const bool   ACC  = (ACCMASK >> o) & 1;      // compile-time after unroll
    const v16h*  wvp  = (const v16h*)args.Wt[o]; // 2 x v16h per transposed row (32B aligned)
    const float* bias = args.bias[o];
    float*       Y    = args.Y[o];

    v8f cres[2];
    #pragma unroll
    for (int tile = 0; tile < 2; ++tile) {
      int col = tile * 16 + ln;
      // B fragment mirrors A with lane -> N: element j = W[16h + j][col] = Wt[col][16h + j]
      v16h b = wvp[col * 2 + half_id];
      float bv = bias[col];
      v8f cacc;
      #pragma unroll
      for (int j = 0; j < 8; ++j) cacc[j] = bv;  // fold bias into C
      cres[tile] = __builtin_amdgcn_wmma_f32_16x16x32_f16(
          false, a, false, b, (short)0, cacc, false, false);
    }

    if (full) {
      // scatter fragments to per-wave LDS tile [row][col], then stream contiguous region
      #pragma unroll
      for (int tile = 0; tile < 2; ++tile) {
        int col = tile * 16 + ln;
        #pragma unroll
        for (int j = 0; j < 8; ++j)
          tl[(half_id * 8 + j) * HID + col] = cres[tile][j];
      }
      __builtin_amdgcn_wave_barrier();           // DS in-order per wave; fence reordering
      float4*       yt = (float4*)(Y + (size_t)r0 * HID);  // 2KB contiguous output tile
      const float4* st = (const float4*)tl;
      #pragma unroll
      for (int i = 0; i < 4; ++i) {
        int idx = i * 32 + lane;
        float4 vv = st[idx];
        if (ACC) {
          float4 yo = yt[idx];
          vv.x += yo.x; vv.y += yo.y; vv.z += yo.z; vv.w += yo.w;
        }
        yt[idx] = vv;
      }
      __builtin_amdgcn_wave_barrier();           // WAR fence before next output reuses tile
    } else {
      #pragma unroll
      for (int tile = 0; tile < 2; ++tile) {
        int col = tile * 16 + ln;
        #pragma unroll
        for (int j = 0; j < 8; ++j) {
          int rowd = r0 + half_id * 8 + j;
          if (rowd < N) {
            float* yp = Y + (size_t)rowd * HID + col;
            if (ACC) *yp += cres[tile][j]; else *yp = cres[tile][j];
          }
        }
      }
    }
  }
}

// s[e] = (1/sqrt(32)) * dot(q[dst], k[src]); atomicMax into m[dst]. One wave per edge.
__global__ void edge_score_kernel(const int* __restrict__ src, const int* __restrict__ dst,
                                  const float* __restrict__ q, const float* __restrict__ k,
                                  float* __restrict__ s, float* __restrict__ m, int E)
{
  int e    = (int)((blockIdx.x * (unsigned)blockDim.x + threadIdx.x) >> 5);
  int lane = threadIdx.x & 31;
  if (e >= E) return;
  int sv = src[e], dv = dst[e];
  float p = q[(size_t)dv * HID + lane] * k[(size_t)sv * HID + lane];
  #pragma unroll
  for (int o = 16; o > 0; o >>= 1) p += __shfl_xor(p, o, 32);
  if (lane == 0) {
    float sc = p * 0.17677669529663687f;
    s[e] = sc;
    atomicMaxF(&m[dv], sc);
  }
}

// w[e] = exp(s[e]-m[dst]); denom[dst] += w. One thread per edge.
__global__ void edge_exp_kernel(const int* __restrict__ dst, float* __restrict__ s,
                                const float* __restrict__ m, float* __restrict__ denom, int E)
{
  int e = blockIdx.x * blockDim.x + threadIdx.x;
  if (e >= E) return;
  int d = dst[e];
  float w = __expf(s[e] - m[d]);
  s[e] = w;
  atomicAdd(&denom[d], w);
}

// acc[dst] += (w/denom[dst]) * v[src]. One wave per edge, lane = channel.
__global__ void edge_scatter_kernel(const int* __restrict__ src, const int* __restrict__ dst,
                                    const float* __restrict__ v, const float* __restrict__ s,
                                    const float* __restrict__ denom, float* __restrict__ acc, int E)
{
  int e    = (int)((blockIdx.x * (unsigned)blockDim.x + threadIdx.x) >> 5);
  int lane = threadIdx.x & 31;
  if (e >= E) return;
  int sv = src[e], dv = dst[e];
  float coeff = s[e] / (denom[dv] + 1e-16f);
  atomicAdd(&acc[(size_t)dv * HID + lane], coeff * v[(size_t)sv * HID + lane]);
}

// Per-channel sum / sumsq of (x*scl). blockDim must be 256 (8 waves). stats pre-zeroed.
__global__ void bn_stats_kernel(const float* __restrict__ x, float* __restrict__ stats,
                                int N, float scl)
{
  int lane = threadIdx.x & 31;
  int wid  = threadIdx.x >> 5;
  int wpb  = blockDim.x >> 5;
  float s = 0.f, s2 = 0.f;
  for (int r = blockIdx.x * wpb + wid; r < N; r += gridDim.x * wpb) {
    float vv = x[(size_t)r * HID + lane] * scl;
    s += vv; s2 += vv * vv;
  }
  __shared__ float sh[2][8][32];
  sh[0][wid][lane] = s; sh[1][wid][lane] = s2;
  __syncthreads();
  if (wid == 0) {
    float t = 0.f, t2 = 0.f;
    #pragma unroll
    for (int w = 0; w < 8; ++w) { t += sh[0][w][lane]; t2 += sh[1][w][lane]; }
    atomicAdd(&stats[lane], t);
    atomicAdd(&stats[32 + lane], t2);
  }
}

// y = LeakyReLU( ((x*scl)-mu)*rsqrt(var+1e-5)*gamma + beta )
__global__ void bn_apply_kernel(const float* __restrict__ x, float* __restrict__ y,
                                const float* __restrict__ stats,
                                const float* __restrict__ gamma, const float* __restrict__ beta,
                                int N, float scl)
{
  int i = blockIdx.x * blockDim.x + threadIdx.x;
  if (i >= N * HID) return;
  int c = i & (HID - 1);
  float invN = 1.f / (float)N;
  float mu   = stats[c] * invN;
  float var  = stats[32 + c] * invN - mu * mu;
  float xn   = (x[i] * scl - mu) * rsqrtf(var + 1e-5f) * gamma[c] + beta[c];
  y[i] = xn > 0.f ? xn : 0.01f * xn;
}

// JumpingKnowledge head: BiLSTM(4 steps, Hl=48) + attention + linear + log_softmax.
// One block per system node, 192 threads (= 4*Hl gates).
__global__ __launch_bounds__(192) void jk_head_kernel(
    const float* __restrict__ seq, int Ns,
    const float* __restrict__ Wi_f, const float* __restrict__ Wh_f, const float* __restrict__ b_f,
    const float* __restrict__ Wi_b, const float* __restrict__ Wh_b, const float* __restrict__ b_b,
    const float* __restrict__ attW, const float* __restrict__ attB,
    const float* __restrict__ linW, const float* __restrict__ linB,
    const float* __restrict__ clsW, const float* __restrict__ clsB,
    float* __restrict__ out)
{
  const int LSEQ = NLAYER + 1, HL = 48, GATES = 4 * HL;
  int n = blockIdx.x;
  int t = threadIdx.x;
  __shared__ float xs[LSEQ][HID];
  __shared__ float h[HL], c[HL], g[GATES];
  __shared__ float hcat[LSEQ][2 * HL];
  __shared__ float alpha[LSEQ];
  __shared__ float xo[HID];

  for (int i = t; i < LSEQ * HID; i += GATES)
    xs[i / HID][i % HID] = seq[(size_t)(i / HID) * Ns * HID + (size_t)n * HID + (i % HID)];
  if (t < HL) { h[t] = 0.f; c[t] = 0.f; }
  __syncthreads();

  // forward direction
  for (int s = 0; s < LSEQ; ++s) {
    float gv = b_f[t];
    for (int kk = 0; kk < HID; ++kk) gv += xs[s][kk] * Wi_f[kk * GATES + t];
    for (int kk = 0; kk < HL; ++kk)  gv += h[kk]    * Wh_f[kk * GATES + t];
    g[t] = gv;
    __syncthreads();
    if (t < HL) {
      float ig = sigm(g[t]), fg = sigm(g[HL + t]);
      float gg = tanhf(g[2 * HL + t]), og = sigm(g[3 * HL + t]);
      float cn = fg * c[t] + ig * gg;
      c[t] = cn;
      float hn = og * tanhf(cn);
      h[t] = hn;
      hcat[s][t] = hn;
    }
    __syncthreads();
  }
  if (t < HL) { h[t] = 0.f; c[t] = 0.f; }
  __syncthreads();
  // backward direction (lax.scan reverse=True: outputs stay time-indexed)
  for (int s = LSEQ - 1; s >= 0; --s) {
    float gv = b_b[t];
    for (int kk = 0; kk < HID; ++kk) gv += xs[s][kk] * Wi_b[kk * GATES + t];
    for (int kk = 0; kk < HL; ++kk)  gv += h[kk]    * Wh_b[kk * GATES + t];
    g[t] = gv;
    __syncthreads();
    if (t < HL) {
      float ig = sigm(g[t]), fg = sigm(g[HL + t]);
      float gg = tanhf(g[2 * HL + t]), og = sigm(g[3 * HL + t]);
      float cn = fg * c[t] + ig * gg;
      c[t] = cn;
      float hn = og * tanhf(cn);
      h[t] = hn;
      hcat[s][HL + t] = hn;
    }
    __syncthreads();
  }
  // attention over layers
  if (t < LSEQ) {
    float a = attB[0];
    for (int j = 0; j < 2 * HL; ++j) a += hcat[t][j] * attW[j];
    alpha[t] = a;
  }
  __syncthreads();
  if (t == 0) {
    float mx = alpha[0];
    for (int s = 1; s < LSEQ; ++s) mx = fmaxf(mx, alpha[s]);
    float se = 0.f;
    for (int s = 0; s < LSEQ; ++s) { alpha[s] = __expf(alpha[s] - mx); se += alpha[s]; }
    for (int s = 0; s < LSEQ; ++s) alpha[s] /= se;
  }
  __syncthreads();
  if (t < HID) {
    float v = 0.f;
    for (int s = 0; s < LSEQ; ++s) v += alpha[s] * xs[s][t];
    xo[t] = v;
  }
  __syncthreads();
  if (t == 0) {
    float r = linB[0];
    float l0 = clsB[0], l1 = clsB[1];
    for (int cc = 0; cc < HID; ++cc) {
      r  += xo[cc] * linW[cc];
      l0 += xo[cc] * clsW[cc * 2 + 0];
      l1 += xo[cc] * clsW[cc * 2 + 1];
    }
    out[n] = r;                                   // x_regr, [Ns,1] flat
    float mx  = fmaxf(l0, l1);
    float lse = mx + __logf(__expf(l0 - mx) + __expf(l1 - mx));
    out[Ns + n * 2 + 0] = l0 - lse;               // xcls, [Ns,2] flat
    out[Ns + n * 2 + 1] = l1 - lse;
  }
}

// ======================== host orchestration ========================
extern "C" void kernel_launch(void* const* d_in, const int* in_sizes, int n_in,
                              void* d_out, int out_size, void* d_ws, size_t ws_size,
                              hipStream_t stream) {
  (void)n_in; (void)out_size; (void)ws_size;
  auto cdiv = [](long a, long b) { return (int)((a + b - 1) / b); };

  enum { ATOM = 0, COORD = 1, MONO = 2, POLY = 3, CPLX = 4, SYS = 5 };
  int Nn[6];
  for (int t = 0; t < 6; ++t) Nn[t] = in_sizes[t] / HID;
  int NMAX = 0; for (int t = 0; t < 6; ++t) if (Nn[t] > NMAX) NMAX = Nn[t];
  int EMAX = 0; for (int i = 6; i < 15; ++i) { int E = in_sizes[i] / 2; if (E > EMAX) EMAX = E; }

  // ---- workspace bump allocator (re-initialized fully every call) ----
  char* wsp = (char*)d_ws;
  auto alloc = [&](size_t bytes) -> void* {
    void* p = (void*)wsp; wsp += (bytes + 255) & ~(size_t)255; return p;
  };
  float* qb    = (float*)alloc((size_t)NMAX * HID * 4);
  float* kb    = (float*)alloc((size_t)NMAX * HID * 4);
  float* vb    = (float*)alloc((size_t)NMAX * HID * 4);
  float* acc   = (float*)alloc((size_t)NMAX * HID * 4);
  float* xa    = (float*)alloc((size_t)Nn[ATOM] * HID * 4);
  float* xm    = (float*)alloc((size_t)Nn[MONO] * HID * 4);
  float* xp    = (float*)alloc((size_t)Nn[POLY] * HID * 4);
  float* xc    = (float*)alloc((size_t)Nn[CPLX] * HID * 4);
  float* xsy   = (float*)alloc((size_t)Nn[SYS] * HID * 4);
  float* hist  = (float*)alloc((size_t)(NLAYER + 1) * Nn[SYS] * HID * 4);
  float* sb    = (float*)alloc((size_t)EMAX * 4);
  float* mb    = (float*)alloc((size_t)NMAX * 4);
  float* db    = (float*)alloc((size_t)NMAX * 4);
  float* stats = (float*)alloc(64 * 4);
  half_t* wh   = (half_t*)alloc((size_t)NLAYER * NREL * 4 * HID * HID * 2);

  // working copies of node features (x_coord is read-only input)
  hipMemcpyAsync(xa,  d_in[0], (size_t)Nn[ATOM] * HID * 4, hipMemcpyDeviceToDevice, stream);
  hipMemcpyAsync(xm,  d_in[2], (size_t)Nn[MONO] * HID * 4, hipMemcpyDeviceToDevice, stream);
  hipMemcpyAsync(xp,  d_in[3], (size_t)Nn[POLY] * HID * 4, hipMemcpyDeviceToDevice, stream);
  hipMemcpyAsync(xc,  d_in[4], (size_t)Nn[CPLX] * HID * 4, hipMemcpyDeviceToDevice, stream);
  hipMemcpyAsync(xsy, d_in[5], (size_t)Nn[SYS]  * HID * 4, hipMemcpyDeviceToDevice, stream);
  hipMemcpyAsync(hist, d_in[5], (size_t)Nn[SYS] * HID * 4, hipMemcpyDeviceToDevice, stream);
  const float* cur[6] = { xa, (const float*)d_in[1], xm, xp, xc, xsy };

  // ---- param leaf indexing (JAX pytree: dict keys sorted alphabetically) ----
  // 15 att.W, 16 att.b, 17 bn.beta, 18 bn.gamma,
  // 19..234 layers[l][rel(alpha)]{Wk,Wq,Ws,Wv,bk,bq,bs,bv},
  // 235 lin.W, 236 lin.b, 237 lincls.W, 238 lincls.b,
  // 239 Wh_b, 240 Wh_f, 241 Wi_b, 242 Wi_f, 243 b_b, 244 b_f
  const int LB = 19;
  auto leaf = [&](int l, int ra, int j) -> const float* {
    return (const float*)d_in[LB + l * 72 + ra * 8 + j];
  };

  // convert all 108 weight matrices to transposed f16 in one launch
  WPtrs wp;
  const int jmap[4] = { 1, 0, 3, 2 };  // my order Wq,Wk,Wv,Ws -> sorted leaves Wk,Wq,Ws,Wv
  for (int l = 0; l < NLAYER; ++l)
    for (int ra = 0; ra < NREL; ++ra)
      for (int m2 = 0; m2 < 4; ++m2)
        wp.p[(l * NREL + ra) * 4 + m2] = leaf(l, ra, jmap[m2]);
  convert_weights_kernel<<<NLAYER * NREL * 4, HID * HID, 0, stream>>>(wp, wh);

  auto fill = [&](float* p, long n, float v) {
    fill_kernel<<<cdiv(n, 256), 256, 0, stream>>>(p, v, (int)n);
  };

  auto tconv = [&](int l, int ra, int ein, const float* XS, int nS,
                   const float* XD, int nD, float* accB) {
    const int* ep = (const int*)d_in[ein];
    int E = in_sizes[ein] / 2;
    const int* esrc = ep;
    const int* edst = ep + E;
    const half_t* Wq = wh + (size_t)((l * NREL + ra) * 4 + 0) * HID * HID;
    const half_t* Wk = Wq + HID * HID;
    const half_t* Wv = Wk + HID * HID;
    const half_t* Ws = Wv + HID * HID;
    const float* bq = leaf(l, ra, 5), *bk = leaf(l, ra, 4);
    const float* bv = leaf(l, ra, 7), *bs = leaf(l, ra, 6);

    if (XS == XD) {
      // self relation: one A read produces q, k, v and the accumulated root term
      LinArgs la;
      la.Wt[0] = Wq; la.bias[0] = bq; la.Y[0] = qb;
      la.Wt[1] = Wk; la.bias[1] = bk; la.Y[1] = kb;
      la.Wt[2] = Wv; la.bias[2] = bv; la.Y[2] = vb;
      la.Wt[3] = Ws; la.bias[3] = bs; la.Y[3] = accB;   // ACC
      int blocks = cdiv(cdiv(nD, 16), 8);
      linear32_wmma_kernel<4, 8><<<blocks, 256, 0, stream>>>(XD, la, nD);
    } else {
      LinArgs kv;                                        // k, v from x_src
      kv.Wt[0] = Wk; kv.bias[0] = bk; kv.Y[0] = kb;
      kv.Wt[1] = Wv; kv.bias[1] = bv; kv.Y[1] = vb;
      kv.Wt[2] = kv.Wt[3] = Wk; kv.bias[2] = kv.bias[3] = bk; kv.Y[2] = kv.Y[3] = kb;
      linear32_wmma_kernel<2, 0><<<cdiv(cdiv(nS, 16), 8), 256, 0, stream>>>(XS, kv, nS);
      LinArgs qs;                                        // q (store) + root (acc) from x_dst
      qs.Wt[0] = Wq; qs.bias[0] = bq; qs.Y[0] = qb;
      qs.Wt[1] = Ws; qs.bias[1] = bs; qs.Y[1] = accB;    // ACC
      qs.Wt[2] = qs.Wt[3] = Wq; qs.bias[2] = qs.bias[3] = bq; qs.Y[2] = qs.Y[3] = qb;
      linear32_wmma_kernel<2, 2><<<cdiv(cdiv(nD, 16), 8), 256, 0, stream>>>(XD, qs, nD);
    }

    fill(mb, nD, -INFINITY);
    fill(db, nD, 0.f);
    edge_score_kernel  <<<cdiv((long)E * 32, 256), 256, 0, stream>>>(esrc, edst, qb, kb, sb, mb, E);
    edge_exp_kernel    <<<cdiv((long)E,       256), 256, 0, stream>>>(edst, sb, mb, db, E);
    edge_scatter_kernel<<<cdiv((long)E * 32, 256), 256, 0, stream>>>(esrc, edst, vb, sb, db, accB, E);
  };

  const float* gamma = (const float*)d_in[18];
  const float* beta  = (const float*)d_in[17];
  auto bnap = [&](float* accB, int N, float scl, float* dest) {
    fill(stats, 64, 0.f);
    int blocks = cdiv(N, 8); if (blocks > 1024) blocks = 1024;
    bn_stats_kernel<<<blocks, 256, 0, stream>>>(accB, stats, N, scl);
    bn_apply_kernel<<<cdiv((long)N * HID, 256), 256, 0, stream>>>(accB, dest, stats, gamma, beta, N, scl);
  };

  // edge array d_in idx: bond=6 polar_aa=7 coord_atom=8 atom_mono=9 chains=10
  //                      polar_mm=11 mono_poly=12 poly_cplx=13 cplx_sys=14
  // alphabetical rel idx: atom_mono=0 bond=1 chains=2 cplx_sys=3 coord_atom=4
  //                       mono_poly=5 polar_aa=6 polar_mm=7 poly_cplx=8
  for (int l = 0; l < NLAYER; ++l) {
    // ---- atom: mean of 3 relations, then BN+LeakyReLU ----
    fill(acc, (long)Nn[ATOM] * HID, 0.f);
    tconv(l, 1, 6, cur[ATOM],  Nn[ATOM],  cur[ATOM], Nn[ATOM], acc);   // bond
    tconv(l, 6, 7, cur[ATOM],  Nn[ATOM],  cur[ATOM], Nn[ATOM], acc);   // polar_aa
    tconv(l, 4, 8, cur[COORD], Nn[COORD], cur[ATOM], Nn[ATOM], acc);   // coord_atom
    bnap(acc, Nn[ATOM], 1.f / 3.f, xa);
    // ---- monomer (uses updated atoms) ----
    fill(acc, (long)Nn[MONO] * HID, 0.f);
    tconv(l, 0, 9,  cur[ATOM], Nn[ATOM], cur[MONO], Nn[MONO], acc);    // atom_mono
    tconv(l, 2, 10, cur[MONO], Nn[MONO], cur[MONO], Nn[MONO], acc);    // chains
    tconv(l, 7, 11, cur[MONO], Nn[MONO], cur[MONO], Nn[MONO], acc);    // polar_mm
    bnap(acc, Nn[MONO], 1.f / 3.f, xm);
    // ---- polymer ----
    fill(acc, (long)Nn[POLY] * HID, 0.f);
    tconv(l, 5, 12, cur[MONO], Nn[MONO], cur[POLY], Nn[POLY], acc);    // mono_poly
    bnap(acc, Nn[POLY], 1.f, xp);
    // ---- complex ----
    fill(acc, (long)Nn[CPLX] * HID, 0.f);
    tconv(l, 8, 13, cur[POLY], Nn[POLY], cur[CPLX], Nn[CPLX], acc);    // poly_cplx
    bnap(acc, Nn[CPLX], 1.f, xc);
    // ---- system ----
    fill(acc, (long)Nn[SYS] * HID, 0.f);
    tconv(l, 3, 14, cur[CPLX], Nn[CPLX], cur[SYS], Nn[SYS], acc);      // cplx_sys
    bnap(acc, Nn[SYS], 1.f, xsy);
    hipMemcpyAsync(hist + (size_t)(l + 1) * Nn[SYS] * HID, xsy,
                   (size_t)Nn[SYS] * HID * 4, hipMemcpyDeviceToDevice, stream);
  }

  // ---- JK BiLSTM + attention + heads ----
  jk_head_kernel<<<Nn[SYS], 192, 0, stream>>>(
      hist, Nn[SYS],
      (const float*)d_in[242], (const float*)d_in[240], (const float*)d_in[244],  // Wi_f, Wh_f, b_f
      (const float*)d_in[241], (const float*)d_in[239], (const float*)d_in[243],  // Wi_b, Wh_b, b_b
      (const float*)d_in[15],  (const float*)d_in[16],                            // att W,b
      (const float*)d_in[235], (const float*)d_in[236],                           // lin W,b
      (const float*)d_in[237], (const float*)d_in[238],                           // lincls W,b
      (float*)d_out);
}
